// Head_89266600280799
// MI455X (gfx1250) — matrix operand order repaired and checked by reading, not compiled, not run on hardware
//
#include <hip/hip_runtime.h>
#include <hip/hip_bf16.h>

// ---------------------------------------------------------------------------
// Causal single-head attention (B=4, T=4096, C=1024, H=64) for gfx1250.
// - All matmuls: v_wmma_f32_16x16x32_bf16 (f32 accumulation).
// - Flash attention with S^T trick (softmax nearly intra-lane; P^T re-enters
//   the PV WMMA as B operand via one 4-dword lane exchange).
// - Double-buffered GLOBAL_LOAD_ASYNC_TO_LDS_B128 staging (ASYNCcnt).
// - Split-K (flash-decoding) partials + combine for parallelism/balance,
//   sized from ws_size on the host.
// ---------------------------------------------------------------------------

typedef __bf16 bf16_t;
typedef __attribute__((ext_vector_type(16))) __bf16 v16bf;
typedef __attribute__((ext_vector_type(8)))  float  v8f;
typedef __attribute__((ext_vector_type(8)))  __bf16 bf16x8;

#define NB 4
#define NT 4096
#define NC 1024
#define NH 64

#define WMMA_BF16(A, B, C) \
  __builtin_amdgcn_wmma_f32_16x16x32_bf16(false, (A), false, (B), (short)0, (C), false, false)

// Async copy of one 16-byte chunk: global -> LDS, tracked by ASYNCcnt.
static __device__ __forceinline__ void async_b128(void* lds, const void* gptr) {
  unsigned ldsoff = (unsigned)(size_t)lds;          // low 32 bits = LDS offset
  asm volatile("global_load_async_to_lds_b128 %0, %1, off"
               :: "v"(ldsoff), "v"(gptr) : "memory");
}
static __device__ __forceinline__ void wait_async0() {
  asm volatile("s_wait_asynccnt 0x0" ::: "memory");
}

// Load a 16-half fragment as two 16-byte chunks (works for A and B patterns).
static __device__ __forceinline__ v16bf ld_frag16(const bf16_t* p0, const bf16_t* p1) {
  union { v16bf v; bf16x8 h[2]; } u;
  u.h[0] = *(const bf16x8*)p0;
  u.h[1] = *(const bf16x8*)p1;
  return u.v;
}

// ---------------------------------------------------------------------------
// Kernel 1: W [C][H] f32 -> WT [H][C] bf16 (Wq pre-scaled by H^-0.5).
// ---------------------------------------------------------------------------
__global__ __launch_bounds__(256) void prep_weights(
    const float* __restrict__ Wk, const float* __restrict__ Wq,
    const float* __restrict__ Wv,
    bf16_t* __restrict__ WkT, bf16_t* __restrict__ WqT, bf16_t* __restrict__ WvT) {
  int idx = blockIdx.x * blockDim.x + threadIdx.x;   // 0 .. NH*NC-1, [h][c]
  int h = idx >> 10;
  int c = idx & (NC - 1);
  WkT[idx] = (bf16_t)Wk[c * NH + h];
  WqT[idx] = (bf16_t)(Wq[c * NH + h] * 0.125f);      // 1/sqrt(64) folded in
  WvT[idx] = (bf16_t)Wv[c * NH + h];
}

// ---------------------------------------------------------------------------
// Kernel 2: projections. One block (4 waves) computes a 16-row tile of all of
// q,k,v; wave w owns output columns [16w,16w+16). x async double-buffered in
// 128-wide k super-steps; W B-fragments software-pipelined one k-step ahead
// so their (L2-resident) latency hides behind the WMMAs.
// Outputs: Qb,Kb as [B*T][64] bf16; V transposed -> VT [B][64][T] bf16.
// ---------------------------------------------------------------------------
__global__ __launch_bounds__(128) void proj_kernel(
    const float* __restrict__ x,
    const bf16_t* __restrict__ WqT, const bf16_t* __restrict__ WkT,
    const bf16_t* __restrict__ WvT,
    bf16_t* __restrict__ Qb, bf16_t* __restrict__ Kb, bf16_t* __restrict__ VT) {
  __shared__ float xs[2][16 * 128];

  const int tid  = threadIdx.x;
  const int wave = tid >> 5;
  const int lane = tid & 31;
  const int col  = lane & 15;
  const int hl   = lane >> 4;
  const int c8   = hl * 8;      // C-layout row base (rows c8..c8+7)
  const int k16  = hl * 16;     // B-fragment K base
  const int row0 = blockIdx.x * 16;          // flat row in [0, B*T)

  // Async prefetch of x[row0..+15][128*ss .. +127] into xs[buf] (4 chunks/thr).
  auto prefetch_x = [&](int ss, int buf) {
#pragma unroll
    for (int it = 0; it < 4; ++it) {
      int ch = it * 128 + tid;                 // 0..511 (16B chunks)
      int r  = ch >> 5;                        // 32 chunks per 128-float row
      int c4 = (ch & 31) * 4;
      async_b128(&xs[buf][r * 128 + c4],
                 &x[(size_t)(row0 + r) * NC + ss * 128 + c4]);
    }
  };

  const int hrow = wave * 16 + col;
  const bf16_t* wqr = &WqT[hrow * NC + k16];
  const bf16_t* wkr = &WkT[hrow * NC + k16];
  const bf16_t* wvr = &WvT[hrow * NC + k16];

  v8f accQ = {}, accK = {}, accV = {};
  prefetch_x(0, 0);

  // Pre-issue B fragments for k-step 0 (pipeline depth 1).
  v16bf bfq = ld_frag16(wqr, wqr + 8);
  v16bf bfk = ld_frag16(wkr, wkr + 8);
  v16bf bfv = ld_frag16(wvr, wvr + 8);

  for (int ss = 0; ss < NC / 128; ++ss) {
    wait_async0();
    __syncthreads();                 // xs[ss&1] ready; xs[(ss+1)&1] now free
    if (ss + 1 < NC / 128) prefetch_x(ss + 1, (ss + 1) & 1);

    const float* xbase = &xs[ss & 1][col * 128];
#pragma unroll
    for (int kk = 0; kk < 4; ++kk) {
      const int k = ss * 128 + kk * 32;
      // A fragment (x rows, bf16), identical for all 4 waves.
      const float* src = xbase + kk * 32;
      float xv[16];
      *(float4*)&xv[0]  = *(const float4*)(src + k16);
      *(float4*)&xv[4]  = *(const float4*)(src + k16 + 4);
      *(float4*)&xv[8]  = *(const float4*)(src + k16 + 8);
      *(float4*)&xv[12] = *(const float4*)(src + k16 + 12);
      v16bf af;
#pragma unroll
      for (int j = 0; j < 16; ++j) af[j] = (bf16_t)xv[j];

      // Issue next k-step's W fragments before consuming the current ones.
      const int kn = k + 32;
      v16bf nq, nk, nv;
      if (kn < NC) {
        nq = ld_frag16(wqr + kn, wqr + kn + 8);
        nk = ld_frag16(wkr + kn, wkr + kn + 8);
        nv = ld_frag16(wvr + kn, wvr + kn + 8);
      }
      accQ = WMMA_BF16(af, bfq, accQ);
      accK = WMMA_BF16(af, bfk, accK);
      accV = WMMA_BF16(af, bfv, accV);
      if (kn < NC) { bfq = nq; bfk = nk; bfv = nv; }
    }
    __syncthreads();                 // done reading xs[ss&1] before refill
  }

  // C layout: lane holds column h, rows = j + 8*hl.
  const int h = hrow;
#pragma unroll
  for (int j = 0; j < 8; ++j) {
    size_t r = (size_t)(row0 + c8 + j);
    Qb[r * NH + h] = (bf16_t)accQ[j];
    Kb[r * NH + h] = (bf16_t)accK[j];
  }
  // V transposed: rows j are consecutive t for fixed h -> one 16B store.
  {
    const int b    = row0 >> 12;
    const int tloc = row0 & (NT - 1);
    bf16x8 vp;
#pragma unroll
    for (int j = 0; j < 8; ++j) vp[j] = (bf16_t)accV[j];
    *(bf16x8*)&VT[(size_t)b * NH * NT + (size_t)h * NT + tloc + c8] = vp;
  }
}

// ---------------------------------------------------------------------------
// Kernel 3: flash attention core. Block = 4 waves = 64 queries; wave owns 16.
// SPLIT=true: processes keys [split*kchunk, min(+kchunk, causal end)) and
// writes unnormalized partials (O, m, l). SPLIT=false: full range, writes out.
// ---------------------------------------------------------------------------
template <bool SPLIT>
__global__ __launch_bounds__(128) void attn_core(
    const bf16_t* __restrict__ Qb, const bf16_t* __restrict__ Kb,
    const bf16_t* __restrict__ VT, float* __restrict__ out,
    float* __restrict__ Opart, float* __restrict__ Mpart,
    float* __restrict__ Lpart, int kchunk) {
  __shared__ bf16_t Ks[2][128 * 64];   // [key][head]
  __shared__ bf16_t Vs[2][64 * 128];   // [head][key]

  const int tid  = threadIdx.x;
  const int wave = tid >> 5;
  const int lane = tid & 31;
  const int col  = lane & 15;
  const int hl   = lane >> 4;
  const int c8   = hl * 8;
  const int k16  = hl * 16;
  const int b    = blockIdx.y;
  const int qblk = blockIdx.x;               // 64-query block
  const int split = SPLIT ? blockIdx.z : 0;
  const int qwave = qblk * 64 + wave * 16;   // this wave's first query
  const int q     = qwave + col;             // this lane's query column

  const int kbeg = SPLIT ? split * kchunk : 0;
  const int kcap = SPLIT ? kbeg + kchunk : NT;
  const int kval = qblk * 64 + 64;           // causal end (exclusive)
  const int kend = kcap < kval ? kcap : kval;
  const int nst  = kend > kbeg ? ((kend - kbeg + 127) >> 7) : 0;

  // Async prefetch of 128 keys of K and V^T into buffers (16 chunks/thread).
  auto prefetch_kv = [&](int kstart, int buf) {
    const bf16_t* ksrc = &Kb[((size_t)b * NT + kstart) * NH];
#pragma unroll
    for (int it = 0; it < 8; ++it) {
      int ch = it * 128 + tid;               // 16B chunk id, linear copy
      async_b128(&Ks[buf][ch * 8], &ksrc[ch * 8]);
    }
#pragma unroll
    for (int it = 0; it < 8; ++it) {
      int ch  = it * 128 + tid;
      int h   = ch >> 4;
      int off = (ch & 15) * 8;
      async_b128(&Vs[buf][h * 128 + off],
                 &VT[(size_t)b * NH * NT + (size_t)h * NT + kstart + off]);
    }
  };

  // Q^T B-fragments (head is the K dimension): contiguous from Qb row.
  const bf16_t* qrow = &Qb[((size_t)b * NT + q) * NH];
  const v16bf qf0 = ld_frag16(qrow + c8,      qrow + 16 + c8);
  const v16bf qf1 = ld_frag16(qrow + 32 + c8, qrow + 48 + c8);

  v8f O0 = {}, O1 = {}, O2 = {}, O3 = {};
  float m_run = -1e30f, l_run = 0.0f;

  if (nst > 0) prefetch_kv(kbeg, 0);

  for (int s = 0; s < nst; ++s) {
    const int kstart = kbeg + (s << 7);
    const int buf = s & 1;
    wait_async0();
    __syncthreads();                 // stage s ready; buffer (s+1)&1 now free
    if (s + 1 < nst) prefetch_kv(kstart + 128, (s + 1) & 1);

#pragma unroll 1
    for (int p = 0; p < 4; ++p) {            // 32-key pair of sub-tiles
      const int kp = kstart + p * 32;
      if (kp > qwave + 15) break;            // uniform causal skip

      // --- S^T sub-tiles (16 keys x 16 queries each) ---
      v8f s0 = {}, s1 = {};
      {
        const bf16_t* kr0 = &Ks[buf][(p * 32 + col) * 64];
        const bf16_t* kr1 = kr0 + 16 * 64;
        v16bf a0 = ld_frag16(kr0 + c8,      kr0 + 16 + c8);
        v16bf a1 = ld_frag16(kr0 + 32 + c8, kr0 + 48 + c8);
        s0 = WMMA_BF16(a0, qf0, s0);
        s0 = WMMA_BF16(a1, qf1, s0);
        v16bf a2 = ld_frag16(kr1 + c8,      kr1 + 16 + c8);
        v16bf a3 = ld_frag16(kr1 + 32 + c8, kr1 + 48 + c8);
        s1 = WMMA_BF16(a2, qf0, s1);
        s1 = WMMA_BF16(a3, qf1, s1);
      }

      // --- causal mask (key row = kp [+16] + j + 8*hl, query col = q) ---
      if (kp + 15 > qwave) {
#pragma unroll
        for (int j = 0; j < 8; ++j)
          if (kp + c8 + j > q) s0[j] = -1e30f;
      }
      if (kp + 31 > qwave) {
#pragma unroll
        for (int j = 0; j < 8; ++j)
          if (kp + 16 + c8 + j > q) s1[j] = -1e30f;
      }

      // --- online softmax (per query column) ---
      float mb = -1e30f;
#pragma unroll
      for (int j = 0; j < 8; ++j) mb = fmaxf(mb, fmaxf(s0[j], s1[j]));
      mb = fmaxf(mb, __shfl_xor(mb, 16, 32));
      const float m_new = fmaxf(m_run, mb);
      const float al    = __expf(m_run - m_new);

      float ps = 0.0f;
      bf16x8 t0h, t1h;
#pragma unroll
      for (int j = 0; j < 8; ++j) {
        float e = __expf(s0[j] - m_new); ps += e; t0h[j] = (bf16_t)e;
      }
#pragma unroll
      for (int j = 0; j < 8; ++j) {
        float e = __expf(s1[j] - m_new); ps += e; t1h[j] = (bf16_t)e;
      }
      ps += __shfl_xor(ps, 16, 32);
      l_run = l_run * al + ps;
      m_run = m_new;
#pragma unroll
      for (int j = 0; j < 8; ++j) {
        O0[j] *= al; O1[j] *= al; O2[j] *= al; O3[j] *= al;
      }

      // --- repack P^T C-layout -> B-fragment via one 4-dword exchange ---
      union U8 { bf16x8 h; int i[4]; };
      U8 snd; snd.h = hl ? t0h : t1h;
      U8 rcv;
#pragma unroll
      for (int j = 0; j < 4; ++j) rcv.i[j] = __shfl_xor(snd.i[j], 16, 32);
      union { v16bf v; bf16x8 h[2]; } pf;
      pf.h[0] = hl ? rcv.h : t0h;
      pf.h[1] = hl ? t1h : rcv.h;

      // --- O += V^T(heads x 32 keys) * P^T ---
      const bf16_t* vb = &Vs[buf][col * 128 + p * 32];
      O0 = WMMA_BF16(ld_frag16(vb + k16, vb + k16 + 8), pf.v, O0);
      vb += 16 * 128;
      O1 = WMMA_BF16(ld_frag16(vb + k16, vb + k16 + 8), pf.v, O1);
      vb += 16 * 128;
      O2 = WMMA_BF16(ld_frag16(vb + k16, vb + k16 + 8), pf.v, O2);
      vb += 16 * 128;
      O3 = WMMA_BF16(ld_frag16(vb + k16, vb + k16 + 8), pf.v, O3);
    }
    __syncthreads();                 // done reading buffer s&1 before refill
  }

  // --- epilogue ---
  const size_t row = (size_t)b * NT + q;
  const float inv = SPLIT ? 1.0f : (1.0f / l_run);
  float* orow = SPLIT ? &Opart[((size_t)split * NB * NT + row) * NH]
                      : &out[row * NH];
  v8f oo[4] = {O0, O1, O2, O3};
#pragma unroll
  for (int t = 0; t < 4; ++t) {
    float4 lo, hi;
    lo.x = oo[t][0] * inv; lo.y = oo[t][1] * inv;
    lo.z = oo[t][2] * inv; lo.w = oo[t][3] * inv;
    hi.x = oo[t][4] * inv; hi.y = oo[t][5] * inv;
    hi.z = oo[t][6] * inv; hi.w = oo[t][7] * inv;
    *(float4*)(orow + t * 16 + c8)     = lo;
    *(float4*)(orow + t * 16 + c8 + 4) = hi;
  }
  if (SPLIT && hl == 0) {
    Mpart[(size_t)split * NB * NT + row] = m_run;
    Lpart[(size_t)split * NB * NT + row] = l_run;
  }
}

// ---------------------------------------------------------------------------
// Kernel 4: merge split-K partials. One thread per (b, q, 8-head group).
// ---------------------------------------------------------------------------
__global__ __launch_bounds__(256) void attn_combine(
    const float* __restrict__ Opart, const float* __restrict__ Mpart,
    const float* __restrict__ Lpart, float* __restrict__ out, int S) {
  int idx = blockIdx.x * blockDim.x + threadIdx.x;
  int h8 = idx & 7;
  size_t row = (size_t)(idx >> 3);           // in [0, NB*NT)

  float M = -1e30f;
  for (int s = 0; s < S; ++s)
    M = fmaxf(M, Mpart[(size_t)s * NB * NT + row]);

  float L = 0.0f;
  float acc[8] = {};
  for (int s = 0; s < S; ++s) {
    float w = __expf(Mpart[(size_t)s * NB * NT + row] - M);
    L += w * Lpart[(size_t)s * NB * NT + row];
    const float* op = &Opart[((size_t)s * NB * NT + row) * NH + h8 * 8];
    float4 a = *(const float4*)op;
    float4 c = *(const float4*)(op + 4);
    acc[0] += w * a.x; acc[1] += w * a.y; acc[2] += w * a.z; acc[3] += w * a.w;
    acc[4] += w * c.x; acc[5] += w * c.y; acc[6] += w * c.z; acc[7] += w * c.w;
  }
  const float inv = 1.0f / L;
  float* orow = &out[row * NH + h8 * 8];
  float4 o0 = {acc[0] * inv, acc[1] * inv, acc[2] * inv, acc[3] * inv};
  float4 o1 = {acc[4] * inv, acc[5] * inv, acc[6] * inv, acc[7] * inv};
  *(float4*)orow = o0;
  *(float4*)(orow + 4) = o1;
}

// ---------------------------------------------------------------------------
extern "C" void kernel_launch(void* const* d_in, const int* in_sizes, int n_in,
                              void* d_out, int out_size, void* d_ws, size_t ws_size,
                              hipStream_t stream) {
  const float* x  = (const float*)d_in[0];
  const float* Wk = (const float*)d_in[1];
  const float* Wq = (const float*)d_in[2];
  const float* Wv = (const float*)d_in[3];
  float* out = (float*)d_out;

  char* ws = (char*)d_ws;
  const size_t WT_BYTES = (size_t)NH * NC * sizeof(bf16_t);      // 128 KB
  const size_t QK_BYTES = (size_t)NB * NT * NH * sizeof(bf16_t); // 2 MB
  bf16_t* WkT = (bf16_t*)(ws);
  bf16_t* WqT = (bf16_t*)(ws + WT_BYTES);
  bf16_t* WvT = (bf16_t*)(ws + 2 * WT_BYTES);
  bf16_t* Qb  = (bf16_t*)(ws + 3 * WT_BYTES);
  bf16_t* Kb  = (bf16_t*)(ws + 3 * WT_BYTES + QK_BYTES);
  bf16_t* VT  = (bf16_t*)(ws + 3 * WT_BYTES + 2 * QK_BYTES);
  const size_t base = 3 * WT_BYTES + 3 * QK_BYTES;               // ~6.4 MB

  // Pick split-K factor from available workspace (deterministic per run).
  const size_t oBytes  = (size_t)NB * NT * NH * sizeof(float);   // 4 MB/split
  const size_t mlBytes = (size_t)NB * NT * sizeof(float);        // 64 KB/split
  int S = 0;
  for (int cand = 4; cand >= 1; cand >>= 1) {
    if (base + (size_t)cand * (oBytes + 2 * mlBytes) <= ws_size) { S = cand; break; }
  }

  prep_weights<<<dim3((NH * NC) / 256), dim3(256), 0, stream>>>(
      Wk, Wq, Wv, WkT, WqT, WvT);
  proj_kernel<<<dim3((NB * NT) / 16), dim3(128), 0, stream>>>(
      x, WqT, WkT, WvT, Qb, Kb, VT);

  if (S > 0) {
    float* Opart = (float*)(ws + base);
    float* Mpart = (float*)(ws + base + (size_t)S * oBytes);
    float* Lpart = (float*)(ws + base + (size_t)S * oBytes + (size_t)S * mlBytes);
    attn_core<true><<<dim3(NT / 64, NB, S), dim3(128), 0, stream>>>(
        Qb, Kb, VT, nullptr, Opart, Mpart, Lpart, NT / S);
    attn_combine<<<dim3((NB * NT * 8) / 256), dim3(256), 0, stream>>>(
        Opart, Mpart, Lpart, out, S);
  } else {
    attn_core<false><<<dim3(NT / 64, NB), dim3(128), 0, stream>>>(
        Qb, Kb, VT, out, nullptr, nullptr, nullptr, NT);
  }
}
